// Multihead_self_attention_26456998543951
// MI455X (gfx1250) — matrix operand order repaired and checked by reading, not compiled, not run on hardware
//
#include <hip/hip_runtime.h>
#include <hip/hip_bf16.h>
#include <math.h>

// ---------------------------------------------------------------------------
// Multihead causal self-attention, f16 WMMA pipeline for gfx1250 (MI455X).
// B=4, S=2048, D=1024, H=16, DK=64.
// ---------------------------------------------------------------------------

typedef _Float16 half_t;
typedef __attribute__((ext_vector_type(16))) _Float16 v16h;
typedef __attribute__((ext_vector_type(8)))  _Float16 v8h;
typedef __attribute__((ext_vector_type(8)))  float    v8f;
typedef __attribute__((ext_vector_type(4)))  int      v4i;

#define ATT_B   4
#define ATT_S   2048
#define ATT_D   1024
#define ATT_H   16
#define ATT_DK  64
#define ATT_BS  (ATT_B * ATT_S)   // 8192 rows

// gfx1250 async global->LDS copy (ASYNCcnt-tracked) if toolchain declares it.
#if defined(__has_builtin)
#  if __has_builtin(__builtin_amdgcn_global_load_async_to_lds_b128)
#    define ATT_ASYNC_LDS 1
#  endif
#endif
#ifndef ATT_ASYNC_LDS
#  define ATT_ASYNC_LDS 0
#endif

typedef __attribute__((address_space(1))) v4i gv4i;   // global (device) AS
typedef __attribute__((address_space(3))) v4i lv4i;   // LDS AS

__device__ __forceinline__ v16h concat8(v8h lo, v8h hi8) {
    return __builtin_shufflevector(lo, hi8, 0,1,2,3,4,5,6,7,8,9,10,11,12,13,14,15);
}

// Stage 16 bytes global -> LDS (async on CDNA5, register relay otherwise).
__device__ __forceinline__ void stage16(half_t* dst, const half_t* src) {
#if ATT_ASYNC_LDS
    __builtin_amdgcn_global_load_async_to_lds_b128(
        (gv4i*)src, (lv4i*)dst, 0, 0);
#else
    *(v8h*)dst = *(const v8h*)src;
#endif
}

// ---------------------------------------------------------------------------
// fp32 -> fp16 conversion
// ---------------------------------------------------------------------------
__global__ void cvt_f32_to_f16(const float* __restrict__ in,
                               half_t* __restrict__ out, int n) {
    int i = blockIdx.x * blockDim.x + threadIdx.x;
    if (i < n) out[i] = (half_t)in[i];
}

// ---------------------------------------------------------------------------
// Y = X (M x K, f16 row-major) @ W^T (W is N x K, f16 row-major)
// Block = 256 threads (8 waves). Block tile = 256(M) x 64(N); wave tile =
// 32 x 64 (2 A frags x 4 B frags = 8 WMMA accumulators). The 64x32 W slab
// per K-step is shared by all 8 waves, so it is staged into LDS once per
// block (double-buffered, async global->LDS on CDNA5), then read as B
// fragments via ds loads.  grid = (M/256, N/64).
// mode 0: Y f16 head layout   [b][h][s][dk]   (Q and K)
// mode 1: Y f16 head layout T [b][h][dk][s]   (V, pre-transposed for PV WMMA)
// mode 2: Y f32 row-major     [M][N]          (final output projection)
// ---------------------------------------------------------------------------
__global__ __launch_bounds__(256)
void wmma_gemm_xwt(const half_t* __restrict__ X, const half_t* __restrict__ W,
                   void* __restrict__ Yout, int mode) {
    const int K = ATT_D;
    const int lane = threadIdx.x & 31;
    const int wave = threadIdx.x >> 5;
    const int l15  = lane & 15;
    const int hi   = (lane >> 4) & 1;

    const int mbase = blockIdx.x * 256 + wave * 32;
    const int nbase = blockIdx.y * 64;

    __shared__ __align__(32) half_t bTile[2][64 * 32];   // 8 KB, double buffer

    // staging assignment: 64 rows x 32 halves = 256 x 16B chunks
    const int srow = threadIdx.x >> 2;          // 0..63  (W row within slab)
    const int sch  = (threadIdx.x & 3) * 8;     // 0,8,16,24 (half offset)
    const half_t* wsrc = W + (size_t)(nbase + srow) * K + sch;
    half_t* sdst0 = &bTile[0][srow * 32 + sch];
    half_t* sdst1 = &bTile[1][srow * 32 + sch];

    const half_t* xrow0 = X + (size_t)(mbase + l15) * K;
    const half_t* xrow1 = X + (size_t)(mbase + 16 + l15) * K;

    v8f acc[2][4] = {};

    // prefetch first W slab
    stage16(sdst0, wsrc);

    int buf = 0;
    for (int kk = 0; kk < K; kk += 32) {
#if ATT_ASYNC_LDS
        asm volatile("s_wait_asynccnt 0x0" ::: "memory");  // own copies landed
#endif
        __syncthreads();                                   // whole slab landed
        if (kk + 32 < K)                                   // prefetch next slab
            stage16(buf ? sdst0 : sdst1, wsrc + kk + 32);

        // A fragments (per-wave rows, direct from global / L2)
        v16h a0 = concat8(*(const v8h*)(xrow0 + kk + hi * 16),
                          *(const v8h*)(xrow0 + kk + hi * 16 + 8));
        v16h a1 = concat8(*(const v8h*)(xrow1 + kk + hi * 16),
                          *(const v8h*)(xrow1 + kk + hi * 16 + 8));

        const half_t* bt = &bTile[buf][0];
#pragma unroll
        for (int nt = 0; nt < 4; ++nt) {
            // B column n == W row n; K slice hi*16..hi*16+15 (same contiguous k order as A)
            v16h bf = *(const v16h*)(bt + (nt * 16 + l15) * 32 + hi * 16);
            acc[0][nt] = __builtin_amdgcn_wmma_f32_16x16x32_f16(
                false, a0, false, bf, (short)0, acc[0][nt], false, false);
            acc[1][nt] = __builtin_amdgcn_wmma_f32_16x16x32_f16(
                false, a1, false, bf, (short)0, acc[1][nt], false, false);
        }
        buf ^= 1;
    }

    // Store: C/D layout -> lane l<16 rows +0..7, l>=16 rows +8..15
#pragma unroll
    for (int am = 0; am < 2; ++am) {
#pragma unroll
        for (int nt = 0; nt < 4; ++nt) {
            const int n = nbase + nt * 16 + l15;
#pragma unroll
            for (int r = 0; r < 8; ++r) {
                const int m = mbase + am * 16 + hi * 8 + r;
                const float v = acc[am][nt][r];
                if (mode == 2) {
                    ((float*)Yout)[(size_t)m * ATT_D + n] = v;
                } else {
                    const int h  = n >> 6;        // n / DK
                    const int dk = n & 63;        // n % DK
                    const int b  = m >> 11;       // m / S
                    const int s  = m & 2047;      // m % S
                    size_t idx;
                    if (mode == 0)
                        idx = (((size_t)b * ATT_H + h) * ATT_S + s) * ATT_DK + dk;
                    else
                        idx = (((size_t)b * ATT_H + h) * ATT_DK + dk) * ATT_S + s;
                    ((half_t*)Yout)[idx] = (half_t)v;
                }
            }
        }
    }
}

// ---------------------------------------------------------------------------
// Causal flash attention: one wave per (head, 16-query tile).
// Q,K: [b*h][s][dk] f16.  Vt: [b*h][dk][s] f16.  Ah: [b][s][D] f16 (heads merged).
// Per 32-key chunk: 4 score WMMAs (K-dim=DK=64 => 2 steps x 2 key blocks),
// online softmax via 16-lane shfl_xor reductions, P staged through LDS to
// convert C-layout -> A-layout, then 4 PV WMMAs against transposed V.
// grid = (S/16, B*H), block = 32 (one wave).
// ---------------------------------------------------------------------------
__global__ __launch_bounds__(32)
void flash_attn_wmma(const half_t* __restrict__ Qh, const half_t* __restrict__ Kh,
                     const half_t* __restrict__ Vt, half_t* __restrict__ Ah) {
    __shared__ __align__(16) half_t pTile[16 * 32];

    const int lane = threadIdx.x & 31;
    const int l15  = lane & 15;
    const int hi   = (lane >> 4) & 1;
    const int qt   = blockIdx.x;          // query tile (0..127)
    const int bh   = blockIdx.y;          // fused batch*head (0..63)
    const int b    = bh >> 4;
    const int h    = bh & 15;

    // Q tile (16 x 64) as two A fragments (K-steps d0=0, d0=32), loaded once.
    const half_t* qbase = Qh + ((size_t)bh * ATT_S + qt * 16 + l15) * ATT_DK;
    v16h qf[2];
#pragma unroll
    for (int i = 0; i < 2; ++i) {
        v8h lo  = *(const v8h*)(qbase + i * 32 + hi * 16);
        v8h hi8 = *(const v8h*)(qbase + i * 32 + hi * 16 + 8);
        qf[i] = concat8(lo, hi8);
    }

    v8f o[4] = {};
    float mrow[8], lrow[8];
#pragma unroll
    for (int r = 0; r < 8; ++r) { mrow[r] = -1e30f; lrow[r] = 0.0f; }

    const int nchunks = qt / 2 + 1;       // causal: keys <= qt*16+15
    for (int c = 0; c < nchunks; ++c) {
        const int kbase = c * 32;

        // ---- scores: two 16x16 blocks (key sub-tiles t=0,1), K-dim = 64 ----
        v8f sblk[2];
#pragma unroll
        for (int t = 0; t < 2; ++t) {
            v8f s = {};
#pragma unroll
            for (int d0 = 0; d0 < 2; ++d0) {
                // B column j = K row j (contiguous along dk)
                const half_t* kptr = Kh +
                    ((size_t)bh * ATT_S + kbase + t * 16 + l15) * ATT_DK +
                    d0 * 32 + hi * 16;
                v16h bf = *(const v16h*)kptr;
                s = __builtin_amdgcn_wmma_f32_16x16x32_f16(
                    false, qf[d0], false, bf, (short)0, s, false, false);
            }
            sblk[t] = s;
        }

        // ---- online softmax ----
#pragma unroll
        for (int r = 0; r < 8; ++r) {
            const int qi = qt * 16 + hi * 8 + r;
#pragma unroll
            for (int t = 0; t < 2; ++t) {
                const int kj = kbase + t * 16 + l15;
                float v = sblk[t][r] * 0.125f;           // 1/sqrt(DK)
                sblk[t][r] = (kj <= qi) ? v : -1e30f;    // causal mask
            }
            float lm = fmaxf(sblk[0][r], sblk[1][r]);
            lm = fmaxf(lm, __shfl_xor(lm, 1));
            lm = fmaxf(lm, __shfl_xor(lm, 2));
            lm = fmaxf(lm, __shfl_xor(lm, 4));
            lm = fmaxf(lm, __shfl_xor(lm, 8));           // rowmax in half-wave
            const float mnew = fmaxf(mrow[r], lm);

            const float p0 = __expf(sblk[0][r] - mnew);
            const float p1 = __expf(sblk[1][r] - mnew);
            sblk[0][r] = p0; sblk[1][r] = p1;
            float ps = p0 + p1;
            ps += __shfl_xor(ps, 1);
            ps += __shfl_xor(ps, 2);
            ps += __shfl_xor(ps, 4);
            ps += __shfl_xor(ps, 8);                     // rowsum

            const float corr = __expf(mrow[r] - mnew);
            lrow[r] = lrow[r] * corr + ps;
            mrow[r] = mnew;
#pragma unroll
            for (int nt = 0; nt < 4; ++nt) o[nt][r] *= corr;
        }

        // ---- P: C-layout -> A-layout via LDS (16x32 f16 tile) ----
#pragma unroll
        for (int r = 0; r < 8; ++r) {
            const int row = hi * 8 + r;
            pTile[row * 32 + l15]      = (half_t)sblk[0][r];
            pTile[row * 32 + 16 + l15] = (half_t)sblk[1][r];
        }
        asm volatile("s_wait_dscnt 0x0" ::: "memory");   // RAW over LDS, same wave
        v8h plo = *(const v8h*)(pTile + l15 * 32 + hi * 16);
        v8h phi = *(const v8h*)(pTile + l15 * 32 + hi * 16 + 8);
        v16h pf = concat8(plo, phi);

        // ---- O += P @ V_chunk : B column d = Vt row d (contiguous along s) ----
#pragma unroll
        for (int nt = 0; nt < 4; ++nt) {
            const half_t* vptr = Vt +
                ((size_t)bh * ATT_DK + nt * 16 + l15) * ATT_S + kbase + hi * 16;
            v16h bf = *(const v16h*)vptr;
            o[nt] = __builtin_amdgcn_wmma_f32_16x16x32_f16(
                false, pf, false, bf, (short)0, o[nt], false, false);
        }
    }

    // ---- normalize and store merged-head activations [b][s][D] ----
#pragma unroll
    for (int r = 0; r < 8; ++r) {
        const float inv = 1.0f / lrow[r];
        const int srow = qt * 16 + hi * 8 + r;
        const size_t rowoff = ((size_t)b * ATT_S + srow) * ATT_D;
#pragma unroll
        for (int nt = 0; nt < 4; ++nt)
            Ah[rowoff + h * 64 + nt * 16 + l15] = (half_t)(o[nt][r] * inv);
    }
}

// ---------------------------------------------------------------------------
// Host launcher
// ---------------------------------------------------------------------------
extern "C" void kernel_launch(void* const* d_in, const int* in_sizes, int n_in,
                              void* d_out, int out_size, void* d_ws, size_t ws_size,
                              hipStream_t stream) {
    (void)in_sizes; (void)n_in; (void)out_size; (void)ws_size;
    const float* x  = (const float*)d_in[0];
    const float* wq = (const float*)d_in[1];
    const float* wk = (const float*)d_in[2];
    const float* wv = (const float*)d_in[3];
    const float* wo = (const float*)d_in[4];

    const size_t NX = (size_t)ATT_BS * ATT_D;   // 8,388,608
    const size_t NW = (size_t)ATT_D * ATT_D;    // 1,048,576

    half_t* Xh  = (half_t*)d_ws;
    half_t* Wqh = Xh  + NX;
    half_t* Wkh = Wqh + NW;
    half_t* Wvh = Wkh + NW;
    half_t* Woh = Wvh + NW;
    half_t* Qh  = Woh + NW;
    half_t* Kh  = Qh  + NX;
    half_t* Vt  = Kh  + NX;
    half_t* Ah  = Vt  + NX;                     // total ~92.3 MB of workspace

    // fp32 -> f16 conversions
    cvt_f32_to_f16<<<(int)((NX + 255) / 256), 256, 0, stream>>>(x,  Xh,  (int)NX);
    cvt_f32_to_f16<<<(int)((NW + 255) / 256), 256, 0, stream>>>(wq, Wqh, (int)NW);
    cvt_f32_to_f16<<<(int)((NW + 255) / 256), 256, 0, stream>>>(wk, Wkh, (int)NW);
    cvt_f32_to_f16<<<(int)((NW + 255) / 256), 256, 0, stream>>>(wv, Wvh, (int)NW);
    cvt_f32_to_f16<<<(int)((NW + 255) / 256), 256, 0, stream>>>(wo, Woh, (int)NW);

    // Q/K/V projections (V stored transposed per head)
    dim3 gproj(ATT_BS / 256, ATT_D / 64);       // 32 x 16
    wmma_gemm_xwt<<<gproj, 256, 0, stream>>>(Xh, Wqh, Qh, 0);
    wmma_gemm_xwt<<<gproj, 256, 0, stream>>>(Xh, Wkh, Kh, 0);
    wmma_gemm_xwt<<<gproj, 256, 0, stream>>>(Xh, Wvh, Vt, 1);

    // causal flash attention
    dim3 gattn(ATT_S / 16, ATT_B * ATT_H);      // 128 x 64 waves
    flash_attn_wmma<<<gattn, 32, 0, stream>>>(Qh, Kh, Vt, Ah);

    // output projection, fp32 result
    wmma_gemm_xwt<<<gproj, 256, 0, stream>>>(Ah, Woh, d_out, 2);
}
